// MultiHeadAttention_2448131359016
// MI455X (gfx1250) — matrix operand order repaired and checked by reading, not compiled, not run on hardware
//
#include <hip/hip_runtime.h>
#include <hip/hip_bf16.h>

#define B_ 2
#define T_ 2048
#define D_ 1024
#define H_ 16
#define DK_ 64

typedef __attribute__((ext_vector_type(16))) _Float16 v16h;
typedef __attribute__((ext_vector_type(8)))  _Float16 v8h;
typedef __attribute__((ext_vector_type(8)))  float    v8f;
typedef __attribute__((ext_vector_type(4)))  float    v4f;

__device__ __forceinline__ v8h cvt8(const float* __restrict__ p) {
    v4f a = *(const v4f*)p;
    v4f b = *(const v4f*)(p + 4);
    v8h r;
    r[0]=(_Float16)a[0]; r[1]=(_Float16)a[1]; r[2]=(_Float16)a[2]; r[3]=(_Float16)a[3];
    r[4]=(_Float16)b[0]; r[5]=(_Float16)b[1]; r[6]=(_Float16)b[2]; r[7]=(_Float16)b[3];
    return r;
}

__device__ __forceinline__ v16h cat16(v8h lo, v8h hi) {
    return __builtin_shufflevector(lo, hi, 0,1,2,3,4,5,6,7,8,9,10,11,12,13,14,15);
}

__device__ __forceinline__ v8f wmma_f16(v16h a, v16h b, v8f c) {
    // D = A(16x32 f16) * B(32x16 f16) + C(16x16 f32)
    return __builtin_amdgcn_wmma_f32_16x16x32_f16(false, a, false, b, (short)0, c,
                                                  false, false);
}

// A-fragment (16x32 f16, row-major): lane (g=l/16,m=l%16) holds row m,
// halves[0..7]=K 8g..8g+7, halves[8..15]=K 16+8g..23+8g   (ISA 7.12.2)
__device__ __forceinline__ v16h load_a_f16(const _Float16* rowp, int g) {
    return cat16(*(const v8h*)(rowp + 8*g), *(const v8h*)(rowp + 16 + 8*g));
}
// B-fragment (32x16 f16, K-contiguous per column): lane holds col m,
// halves[0..7]=K 8g..8g+7, halves[8..15]=K 16+8g..23+8g
__device__ __forceinline__ v16h load_b_f16(const _Float16* colp, int g) {
    return cat16(*(const v8h*)(colp + 8*g), *(const v8h*)(colp + 16 + 8*g));
}

// LDS byte offset of a generic pointer known to point into shared memory
__device__ __forceinline__ uint32_t lds_off(const void* p) {
    return (uint32_t)(uintptr_t)(__attribute__((address_space(3))) const void*)p;
}

// 16-byte async DMA: global -> LDS, tracked by ASYNCcnt (CDNA5 §10 / 08_async_tensor)
__device__ __forceinline__ void async_ld16(uint32_t lds_byte_off, uint64_t gaddr) {
    asm volatile("global_load_async_to_lds_b128 %0, %1, off"
                 :: "v"(lds_byte_off), "v"(gaddr) : "memory");
}

// ---------------------------------------------------------------------------
// Kernel 0: elementwise f32 -> f16 (8 elems/thread, b128 in / b128 out)
// ---------------------------------------------------------------------------
__global__ __launch_bounds__(256) void cvt_f32_to_f16_kernel(
    const float* __restrict__ src, _Float16* __restrict__ dst)
{
    const size_t i = ((size_t)blockIdx.x * 256 + threadIdx.x) * 8;
    *(v8h*)(dst + i) = cvt8(src + i);
}

// ---------------------------------------------------------------------------
// Kernel 1: fused QKV projection (NT gemm, f16 in, f32 acc) + bias + RoPE /
// V-transpose. Each wave owns a 16x64 strip: 1 A frag + 4 B frags -> 4 WMMAs.
// grid (D/256, B*T/16, 3), block 128
// ---------------------------------------------------------------------------
__global__ __launch_bounds__(128) void qkv_rope_kernel(
    const _Float16* __restrict__ qf, const _Float16* __restrict__ kf,
    const _Float16* __restrict__ vf,
    const _Float16* __restrict__ Wqf, const _Float16* __restrict__ Wkf,
    const _Float16* __restrict__ Wvf,
    const float* __restrict__ bq, const float* __restrict__ bk,
    const float* __restrict__ bv,
    const float* __restrict__ cosT, const float* __restrict__ sinT,
    _Float16* __restrict__ Qh, _Float16* __restrict__ Kh, _Float16* __restrict__ Vt)
{
    const int z = blockIdx.z;
    const _Float16 *X, *W;
    const float *bias;
    if (z == 0)      { X = qf; W = Wqf; bias = bq; }
    else if (z == 1) { X = kf; W = Wkf; bias = bk; }
    else             { X = vf; W = Wvf; bias = bv; }

    const int lane = threadIdx.x & 31, w = threadIdx.x >> 5;
    const int g = lane >> 4, m = lane & 15;
    const int n0 = blockIdx.x * 256 + w * 64;  // 64-wide output strip
    const int t0 = blockIdx.y * 16;            // flattened B*T row tile

    const _Float16* arow = X + (size_t)(t0 + m) * D_;
    const _Float16* b0r  = W + (size_t)(n0      + m) * D_;
    const _Float16* b1r  = W + (size_t)(n0 + 16 + m) * D_;
    const _Float16* b2r  = W + (size_t)(n0 + 32 + m) * D_;
    const _Float16* b3r  = W + (size_t)(n0 + 48 + m) * D_;

    v8f c0 = {0.f,0.f,0.f,0.f,0.f,0.f,0.f,0.f}, c1 = c0, c2 = c0, c3 = c0;
    for (int kk = 0; kk < D_; kk += 32) {
        v16h a = load_a_f16(arow + kk, g);
        c0 = wmma_f16(a, load_b_f16(b0r + kk, g), c0);
        c1 = wmma_f16(a, load_b_f16(b1r + kk, g), c1);
        c2 = wmma_f16(a, load_b_f16(b2r + kk, g), c2);
        c3 = wmma_f16(a, load_b_f16(b3r + kk, g), c3);
    }

    v8f acc[4] = {c0, c1, c2, c3};
    #pragma unroll
    for (int j = 0; j < 4; j++) {
        const int n  = n0 + j*16 + m;
        const float bn = bias[n];
        const int h  = n >> 6;     // head
        const int dk = n & 63;     // dim within head
        if (z < 2) {
            const int   i   = dk >> 1;
            const float sgn = (dk & 1) ? 1.f : -1.f;  // even: x1*c - x2*s ; odd: x2*c + x1*s
            _Float16* Out = (z == 0) ? Qh : Kh;
            #pragma unroll
            for (int r = 0; r < 8; r++) {
                const int bt = t0 + r + 8*g;
                const int bi = bt >> 11;            // batch (T=2048)
                const int t  = bt & (T_ - 1);       // seq pos
                float val = acc[j][r] + bn;
                float partner = __shfl_xor(val, 1, 32);  // paired rotary component
                float cv = cosT[t * (DK_/2) + i];
                float sv = sinT[t * (DK_/2) + i];
                Out[((size_t)(bi*H_ + h)*T_ + t)*DK_ + dk] =
                    (_Float16)(val * cv + sgn * partner * sv);
            }
        } else {
            // V stored transposed [B,H,DK,T]: consecutive r -> consecutive t,
            // stores coalesce into contiguous b128 per lane
            #pragma unroll
            for (int r = 0; r < 8; r++) {
                const int bt = t0 + r + 8*g;
                const int bi = bt >> 11;
                const int t  = bt & (T_ - 1);
                Vt[((size_t)(bi*H_ + h)*DK_ + dk)*T_ + t] = (_Float16)(acc[j][r] + bn);
            }
        }
    }
}

// ---------------------------------------------------------------------------
// Kernel 2: causal flash attention. 4 waves/WG share async-DMA'd K/V tiles in
// LDS; uniform trip count so __syncthreads is legal. grid (T/64, H, B), blk 128
// ---------------------------------------------------------------------------
__global__ __launch_bounds__(128) void flash_attn_kernel(
    const _Float16* __restrict__ Qh, const _Float16* __restrict__ Kh,
    const _Float16* __restrict__ Vt, _Float16* __restrict__ Ct)
{
    __shared__ __align__(128) _Float16 sK[32 * DK_];     // 32 keys x 64 dk   (4 KB)
    __shared__ __align__(128) _Float16 sV[DK_ * 32];     // 64 dk  x 32 keys  (4 KB)
    __shared__ __align__(128) _Float16 sP[4][16 * 32];   // per-wave P stage  (4 KB)

    const int tid  = threadIdx.x;
    const int lane = tid & 31, w = tid >> 5;
    const int g = lane >> 4, m = lane & 15;
    const int qt0 = blockIdx.x * 64 + w * 16;
    const int h = blockIdx.y, b = blockIdx.z;

    const _Float16* Qb = Qh + (size_t)(b*H_ + h) * T_ * DK_;
    const _Float16* Kb = Kh + (size_t)(b*H_ + h) * T_ * DK_;
    const _Float16* Vb = Vt + (size_t)(b*H_ + h) * DK_ * T_;

    // Q fragments (dk 0..31, 32..63), pre-scaled by 1/sqrt(64)=0.125 (exact in f16)
    const _Float16* qrow = Qb + (size_t)(qt0 + m) * DK_;
    v16h aq0 = load_a_f16(qrow,      g);
    v16h aq1 = load_a_f16(qrow + 32, g);
    #pragma unroll
    for (int i = 0; i < 16; i++) { aq0[i] *= (_Float16)0.125f; aq1[i] *= (_Float16)0.125f; }

    float mi[8], li[8];
    v8f o0 = {0.f,0.f,0.f,0.f,0.f,0.f,0.f,0.f}, o1 = o0, o2 = o0, o3 = o0;
    #pragma unroll
    for (int r = 0; r < 8; r++) { mi[r] = -1e30f; li[r] = 0.f; }

    const v8f zf = {0.f,0.f,0.f,0.f,0.f,0.f,0.f,0.f};
    // Workgroup-uniform: keys up to blockIdx.x*64+63 (extra blocks for low waves
    // are fully masked -> provable no-op in the online-softmax update)
    const int nkb = 2 * blockIdx.x + 2;

    // Async-copy slices: K tile is 4KB contiguous; V tile is 64 rows x 64B
    const uint32_t sK_off = lds_off(sK), sV_off = lds_off(sV);
    const int vrow0 = tid >> 2,  vc0 = (tid & 3) * 8;          // chunk tid
    const int vrow1 = vrow0 + 32, vc1 = vc0;                   // chunk tid+128

    for (int kb = 0; kb < nkb; kb++) {
        const int k0 = kb * 32;

        __syncthreads();   // previous block's LDS reads complete
        // ---- stage K (32x64) and V (64x32) tiles into LDS via async DMA
        {
            const uint64_t gK = (uint64_t)(uintptr_t)(Kb + (size_t)k0 * DK_);
            async_ld16(sK_off + tid*16,        gK + tid*16);
            async_ld16(sK_off + tid*16 + 2048, gK + tid*16 + 2048);
            const uint64_t gV0 = (uint64_t)(uintptr_t)(Vb + (size_t)vrow0*T_ + k0 + vc0);
            const uint64_t gV1 = (uint64_t)(uintptr_t)(Vb + (size_t)vrow1*T_ + k0 + vc1);
            async_ld16(sV_off + vrow0*64 + vc0*2, gV0);
            async_ld16(sV_off + vrow1*64 + vc1*2, gV1);
        }
        asm volatile("s_wait_asynccnt 0x0" ::: "memory");
        __syncthreads();   // tiles visible to all 4 waves

        // ---- S = Q K^T (two 16-key tiles), 4 WMMAs, B-frags from LDS
        const _Float16* kr0 = sK + m * DK_;          // key k0+m
        const _Float16* kr1 = sK + (16 + m) * DK_;   // key k0+16+m
        v8f s0 = wmma_f16(aq0, load_b_f16(kr0, g), zf);
        s0     = wmma_f16(aq1, load_b_f16(kr0 + 32, g), s0);
        v8f s1 = wmma_f16(aq0, load_b_f16(kr1, g), zf);
        s1     = wmma_f16(aq1, load_b_f16(kr1 + 32, g), s1);

        // ---- causal mask (reference: where(mask==0, -1e9))
        const int key0 = k0 + m, key1 = k0 + 16 + m;
        #pragma unroll
        for (int r = 0; r < 8; r++) {
            const int qr = qt0 + r + 8*g;
            if (key0 > qr) s0[r] = -1e9f;
            if (key1 > qr) s1[r] = -1e9f;
        }

        // ---- online softmax update (butterfly over the 16-lane row group)
        float mx[8];
        #pragma unroll
        for (int r = 0; r < 8; r++) mx[r] = fmaxf(s0[r], s1[r]);
        #pragma unroll
        for (int d = 1; d < 16; d <<= 1)
            #pragma unroll
            for (int r = 0; r < 8; r++) mx[r] = fmaxf(mx[r], __shfl_xor(mx[r], d, 32));

        float al[8], p0[8], p1[8], rs[8];
        #pragma unroll
        for (int r = 0; r < 8; r++) {
            float mn = fmaxf(mi[r], mx[r]);
            al[r] = __expf(mi[r] - mn);
            mi[r] = mn;
            p0[r] = __expf(s0[r] - mn);
            p1[r] = __expf(s1[r] - mn);
            rs[r] = p0[r] + p1[r];
        }
        #pragma unroll
        for (int d = 1; d < 16; d <<= 1)
            #pragma unroll
            for (int r = 0; r < 8; r++) rs[r] += __shfl_xor(rs[r], d, 32);
        #pragma unroll
        for (int r = 0; r < 8; r++) {
            li[r] = li[r] * al[r] + rs[r];
            o0[r] *= al[r]; o1[r] *= al[r]; o2[r] *= al[r]; o3[r] *= al[r];
        }

        // ---- P (C-layout f32) -> A-fragment f16 via per-wave LDS transpose
        _Float16* Pb = sP[w];
        #pragma unroll
        for (int r = 0; r < 8; r++) {
            const int row = r + 8*g;
            Pb[row*32 + m]      = (_Float16)p0[r];
            Pb[row*32 + 16 + m] = (_Float16)p1[r];
        }
        asm volatile("s_wait_dscnt 0x0" ::: "memory");  // per-wave DScnt: all lanes' stores done
        v16h ap = load_a_f16(Pb + m*32, g);

        // ---- O += P V, 4 WMMAs, B-frags from the LDS V tile
        o0 = wmma_f16(ap, load_b_f16(sV + ( 0 + m)*32, g), o0);
        o1 = wmma_f16(ap, load_b_f16(sV + (16 + m)*32, g), o1);
        o2 = wmma_f16(ap, load_b_f16(sV + (32 + m)*32, g), o2);
        o3 = wmma_f16(ap, load_b_f16(sV + (48 + m)*32, g), o3);
    }

    // ---- normalize and write concat output [B,T,D] f16
    #pragma unroll
    for (int r = 0; r < 8; r++) {
        const float inv = 1.f / li[r];
        const int t = qt0 + r + 8*g;
        _Float16* out = Ct + ((size_t)(b*T_ + t))*D_ + h*DK_;
        out[ 0 + m] = (_Float16)(o0[r] * inv);
        out[16 + m] = (_Float16)(o1[r] * inv);
        out[32 + m] = (_Float16)(o2[r] * inv);
        out[48 + m] = (_Float16)(o3[r] * inv);
    }
}

// ---------------------------------------------------------------------------
// Kernel 3: out = concat @ Wo^T + bo (all-f16 operands, fp32 out)
// Each wave owns a 16x64 strip. grid (D/256, B*T/16), block 128
// ---------------------------------------------------------------------------
__global__ __launch_bounds__(128) void out_proj_kernel(
    const _Float16* __restrict__ Ct, const _Float16* __restrict__ Wof,
    const float* __restrict__ bo, float* __restrict__ out)
{
    const int lane = threadIdx.x & 31, w = threadIdx.x >> 5;
    const int g = lane >> 4, m = lane & 15;
    const int n0 = blockIdx.x * 256 + w * 64;
    const int t0 = blockIdx.y * 16;

    const _Float16* arow = Ct  + (size_t)(t0 + m) * D_;
    const _Float16* b0r  = Wof + (size_t)(n0      + m) * D_;
    const _Float16* b1r  = Wof + (size_t)(n0 + 16 + m) * D_;
    const _Float16* b2r  = Wof + (size_t)(n0 + 32 + m) * D_;
    const _Float16* b3r  = Wof + (size_t)(n0 + 48 + m) * D_;

    v8f c0 = {0.f,0.f,0.f,0.f,0.f,0.f,0.f,0.f}, c1 = c0, c2 = c0, c3 = c0;
    for (int kk = 0; kk < D_; kk += 32) {
        v16h a = load_a_f16(arow + kk, g);
        c0 = wmma_f16(a, load_b_f16(b0r + kk, g), c0);
        c1 = wmma_f16(a, load_b_f16(b1r + kk, g), c1);
        c2 = wmma_f16(a, load_b_f16(b2r + kk, g), c2);
        c3 = wmma_f16(a, load_b_f16(b3r + kk, g), c3);
    }

    v8f acc[4] = {c0, c1, c2, c3};
    #pragma unroll
    for (int j = 0; j < 4; j++) {
        const int n = n0 + j*16 + m;
        const float bn = bo[n];
        #pragma unroll
        for (int r = 0; r < 8; r++)
            out[(size_t)(t0 + r + 8*g)*D_ + n] = acc[j][r] + bn;
    }
}

// ---------------------------------------------------------------------------
extern "C" void kernel_launch(void* const* d_in, const int* in_sizes, int n_in,
                              void* d_out, int out_size, void* d_ws, size_t ws_size,
                              hipStream_t stream) {
    const float* q    = (const float*)d_in[0];
    const float* k    = (const float*)d_in[1];
    const float* v    = (const float*)d_in[2];
    // d_in[3] = mask (tril, int32) — causal mask applied analytically
    const float* Wq   = (const float*)d_in[4];
    const float* bq   = (const float*)d_in[5];
    const float* Wk   = (const float*)d_in[6];
    const float* bk   = (const float*)d_in[7];
    const float* Wv   = (const float*)d_in[8];
    const float* bv   = (const float*)d_in[9];
    const float* Wo   = (const float*)d_in[10];
    const float* bo   = (const float*)d_in[11];
    const float* cosT = (const float*)d_in[12];
    const float* sinT = (const float*)d_in[13];
    float* out = (float*)d_out;

    // Workspace layout (halves): 3 act copies (4M ea) + 4 weight copies (1M ea)
    // + Qh/Kh/Vt/Ct (4M ea) = 32M halves = 64 MB
    const size_t NA = (size_t)B_ * T_ * D_;   // 4,194,304
    const size_t NW = (size_t)D_ * D_;        // 1,048,576
    _Float16* wsp = (_Float16*)d_ws;
    _Float16* qf  = wsp;
    _Float16* kf  = qf  + NA;
    _Float16* vf  = kf  + NA;
    _Float16* Wqf = vf  + NA;
    _Float16* Wkf = Wqf + NW;
    _Float16* Wvf = Wkf + NW;
    _Float16* Wof = Wvf + NW;
    _Float16* Qh  = Wof + NW;
    _Float16* Kh  = Qh  + NA;
    _Float16* Vt  = Kh  + NA;
    _Float16* Ct  = Vt  + NA;

    dim3 blk(128);
    dim3 cb(256);
    // Stage 0: one-time f32 -> f16 conversion (removes cvt from GEMM loops)
    cvt_f32_to_f16_kernel<<<dim3(NA/2048), cb, 0, stream>>>(q,  qf);
    cvt_f32_to_f16_kernel<<<dim3(NA/2048), cb, 0, stream>>>(k,  kf);
    cvt_f32_to_f16_kernel<<<dim3(NA/2048), cb, 0, stream>>>(v,  vf);
    cvt_f32_to_f16_kernel<<<dim3(NW/2048), cb, 0, stream>>>(Wq, Wqf);
    cvt_f32_to_f16_kernel<<<dim3(NW/2048), cb, 0, stream>>>(Wk, Wkf);
    cvt_f32_to_f16_kernel<<<dim3(NW/2048), cb, 0, stream>>>(Wv, Wvf);
    cvt_f32_to_f16_kernel<<<dim3(NW/2048), cb, 0, stream>>>(Wo, Wof);

    qkv_rope_kernel<<<dim3(D_/256, (B_*T_)/16, 3), blk, 0, stream>>>(
        qf, kf, vf, Wqf, Wkf, Wvf, bq, bk, bv, cosT, sinT, Qh, Kh, Vt);
    flash_attn_kernel<<<dim3(T_/64, H_, B_), blk, 0, stream>>>(Qh, Kh, Vt, Ct);
    out_proj_kernel<<<dim3(D_/256, (B_*T_)/16), blk, 0, stream>>>(Ct, Wof, bo, out);
}
